// Model_39676907885462
// MI455X (gfx1250) — matrix operand order repaired and checked by reading, not compiled, not run on hardware
//
#include <hip/hip_runtime.h>

typedef __attribute__((ext_vector_type(16))) __bf16 v16bf;
typedef __attribute__((ext_vector_type(8)))  __bf16 v8bf;
typedef __attribute__((ext_vector_type(2)))  __bf16 v2bf;
typedef __attribute__((ext_vector_type(8)))  float  v8f;

namespace {
constexpr int kB       = 16;
constexpr int kLQ      = 2048;
constexpr int kLKV     = 2048;
constexpr int kD       = 64;
constexpr int kChunk   = 64;              // keys per mainloop iteration
constexpr int kNChunks = kLKV / kChunk;   // 32
constexpr int kQStride = 68;              // Q panel row stride (floats): 272B, bank-spread
constexpr int kKStride = 72;              // K tile row stride (bf16):    144B, bank-spread
constexpr int kVStride = 72;              // V^T row stride (bf16):       144B, bank-spread
constexpr float kLog2e = 1.44269504088896340736f;

__device__ __forceinline__ v16bf cat8(v8bf a, v8bf b) {
  return __builtin_shufflevector(a, b, 0, 1, 2, 3, 4, 5, 6, 7,
                                 8, 9, 10, 11, 12, 13, 14, 15);
}
__device__ __forceinline__ v8bf cvt_f4x2(float4 a, float4 b) {
  v8bf r;
  r[0] = (__bf16)a.x; r[1] = (__bf16)a.y; r[2] = (__bf16)a.z; r[3] = (__bf16)a.w;
  r[4] = (__bf16)b.x; r[5] = (__bf16)b.y; r[6] = (__bf16)b.z; r[7] = (__bf16)b.w;
  return r;
}
__device__ __forceinline__ v8bf cvt_f4x2s(float4 a, float4 b, float s) {
  v8bf r;
  r[0] = (__bf16)(a.x * s); r[1] = (__bf16)(a.y * s);
  r[2] = (__bf16)(a.z * s); r[3] = (__bf16)(a.w * s);
  r[4] = (__bf16)(b.x * s); r[5] = (__bf16)(b.y * s);
  r[6] = (__bf16)(b.z * s); r[7] = (__bf16)(b.w * s);
  return r;
}
// pack two f32 probabilities into one dword of bf16x2 (v_cvt_pk_bf16_f32)
__device__ __forceinline__ unsigned pkbf(float a, float b) {
  v2bf t; t[0] = (__bf16)a; t[1] = (__bf16)b;
  return __builtin_bit_cast(unsigned, t);
}
// Assemble P^T B-operand (32 keys x 16 queries) from two 16-key S^T tiles.
// dest dword v (elems 2v,2v+1), lane n (q=n%16, h=n/16): key = e + 16h.
//   v<4 : h=0 -> own lowtile pair;        h=1 -> hightile pair from lane^16
//   v>=4: h=0 -> lowtile pair from lane^16; h=1 -> own hightile pair
__device__ __forceinline__ v16bf make_pb(const float* plo, const float* phi2,
                                         bool hihalf) {
  union { unsigned u[8]; v16bf v; } out;
#pragma unroll
  for (int v = 0; v < 4; ++v) {
    const unsigned ca = pkbf(plo[2 * v], plo[2 * v + 1]);
    const unsigned cb = pkbf(phi2[2 * v], phi2[2 * v + 1]);
    const unsigned xa = (unsigned)__shfl_xor((int)ca, 16, 32);
    const unsigned xb = (unsigned)__shfl_xor((int)cb, 16, 32);
    out.u[v]     = hihalf ? xb : ca;
    out.u[v + 4] = hihalf ? cb : xa;
  }
  return out.v;
}
} // namespace

// Flash attention, one 16-query tile per wave, full D=64, 64 keys/iteration.
//   S^T = K * Q^T        (A = bf16 K rows from LDS, B = pre-scaled Q columns)
//   O^T += V^T * P^T     (A = bf16 V^T fragments from LDS, B = P^T from S^T)
// Q is pre-scaled by log2e/scale so S^T lands directly in the exp2 domain.
__global__ __launch_bounds__(256, 1) void fattn_bf16_wmma(
    const float* __restrict__ Q, const float* __restrict__ K,
    const float* __restrict__ V, const int* __restrict__ scale_p,
    float* __restrict__ Out)
{
  __shared__ __align__(16) float  Qlds[128 * kQStride];        // 34816 B (f32, async-staged)
  __shared__ __align__(16) __bf16 Kbls[2][kChunk * kKStride];  // 2 x 9216 B (bf16 rows)
  __shared__ __align__(16) __bf16 Vtls[2][kD * kVStride];      // 2 x 9216 B (bf16, transposed)

  const int  tid  = threadIdx.x;
  const int  lane = tid & 31;
  const int  wave = tid >> 5;
  const bool hi   = lane >= 16;
  const int  qcol = lane & 15;

  const int batch = blockIdx.y;
  const int qrow  = blockIdx.x * 128 + wave * 16 + qcol;

  const float ksc = kLog2e / (float)scale_p[0];   // folded into Q fragments

  const float* __restrict__ Qb = Q + (size_t)batch * kLQ  * kD;
  const float* __restrict__ Kb = K + (size_t)batch * kLKV * kD;
  const float* __restrict__ Vb = V + (size_t)batch * kLKV * kD;

  // ---- stage helpers: thread owns key row = tid&63, 16-wide d-block ----------
  const int srow  = tid & 63;
  const int sdblk = (tid >> 6) * 16;
  auto stage_load = [&](int kc, float4 (&kf)[4], float4 (&vf)[4]) {
    const float4* kp =
        reinterpret_cast<const float4*>(Kb + (size_t)(kc + srow) * kD + sdblk);
    kf[0] = kp[0]; kf[1] = kp[1]; kf[2] = kp[2]; kf[3] = kp[3];
    const float4* vp =
        reinterpret_cast<const float4*>(Vb + (size_t)(kc + srow) * kD + sdblk);
    vf[0] = vp[0]; vf[1] = vp[1]; vf[2] = vp[2]; vf[3] = vp[3];
  };
  auto stage_store = [&](int buf, const float4 (&kf)[4], const float4 (&vf)[4]) {
    // K: bf16 row-major, 16 contiguous elems -> two ds_store_b128
    __bf16* krow = &Kbls[buf][srow * kKStride + sdblk];
    *reinterpret_cast<v8bf*>(krow)     = cvt_f4x2(kf[0], kf[1]);
    *reinterpret_cast<v8bf*>(krow + 8) = cvt_f4x2(kf[2], kf[3]);
    // V: bf16 transposed (Vt[d][key])
    __bf16* vt = &Vtls[buf][0];
    const float vv[16] = {vf[0].x, vf[0].y, vf[0].z, vf[0].w,
                          vf[1].x, vf[1].y, vf[1].z, vf[1].w,
                          vf[2].x, vf[2].y, vf[2].z, vf[2].w,
                          vf[3].x, vf[3].y, vf[3].z, vf[3].w};
#pragma unroll
    for (int j = 0; j < 16; ++j)
      vt[(sdblk + j) * kVStride + srow] = (__bf16)vv[j];
  };

  // ---- prologue: async-stage the block's Q panel (128 rows x 256B) to LDS,
  // rows padded to 272B. 2048 x 16B transfers tracked by ASYNCcnt, 8/thread.
  {
    const unsigned qoff0 = (unsigned)(uintptr_t)(&Qlds[0]);
    const float* qg = Qb + (size_t)(blockIdx.x * 128) * kD;
#pragma unroll
    for (int j = 0; j < 8; ++j) {
      const int x   = tid + 256 * j;      // 0..2047
      const int row = x >> 4;
      const int seg = x & 15;
      const unsigned loff = qoff0 + (unsigned)(row * (kQStride * 4) + seg * 16);
      const unsigned long long ga =
          (unsigned long long)(qg + (size_t)row * kD + seg * 4);
      asm volatile("global_load_async_to_lds_b128 %0, %1, off"
                   :: "v"(loff), "v"(ga) : "memory");
    }
  }
  {
    float4 kf[4], vf[4];
    stage_load(0, kf, vf);
    stage_store(0, kf, vf);
  }
  asm volatile("s_wait_asynccnt 0" ::: "memory");
  __syncthreads();   // Q panel + chunk-0 staging visible to all waves

  // ---- Q as B-operand of S^T (from LDS, pre-scaled by ksc; lane = query col).
  // B elem e <-> d = dchunk*32 + e + 16*(lane>=16)
  v16bf qb0, qb1;
  {
    const float* qp = &Qlds[(wave * 16 + qcol) * kQStride + (hi ? 16 : 0)];
    const float4* p0 = reinterpret_cast<const float4*>(qp);
    qb0 = cat8(cvt_f4x2s(p0[0], p0[1], ksc), cvt_f4x2s(p0[2], p0[3], ksc));
    const float4* p1 = reinterpret_cast<const float4*>(qp + 32);
    qb1 = cat8(cvt_f4x2s(p1[0], p1[1], ksc), cvt_f4x2s(p1[2], p1[3], ksc));
  }

  v8f   acc[4] = {v8f{}, v8f{}, v8f{}, v8f{}};  // O^T: d-tiles 0..3
  float m_run  = -__builtin_inff();             // running max (exp2 domain)
  float l_run  = 0.0f;

  for (int i = 0; i < kNChunks; ++i) {
    const int  buf  = i & 1;
    const bool more = (i + 1) < kNChunks;

    // issue next chunk's global loads before compute (latency hidden under WMMA)
    float4 nk[4], nv[4];
    if (more) stage_load((i + 1) * kChunk, nk, nv);

    // ---- S^T tiles: four 16-key tiles, bf16 K fragments from LDS.
    // A elem e <-> d = dchunk*32 + e + 16*(lane>=16); row = key
    v8f st[4];
#pragma unroll
    for (int t = 0; t < 4; ++t) {
      const __bf16* kr = &Kbls[buf][(t * 16 + qcol) * kKStride + (hi ? 16 : 0)];
      const v16bf ka0 = cat8(*reinterpret_cast<const v8bf*>(kr),
                             *reinterpret_cast<const v8bf*>(kr + 8));
      v8f c = {};
      c = __builtin_amdgcn_wmma_f32_16x16x32_bf16(false, ka0, false, qb0,
                                                  (short)0, c, false, false);
      const v16bf ka1 = cat8(*reinterpret_cast<const v8bf*>(kr + 32),
                             *reinterpret_cast<const v8bf*>(kr + 40));
      c = __builtin_amdgcn_wmma_f32_16x16x32_bf16(false, ka1, false, qb1,
                                                  (short)0, c, false, false);
      st[t] = c;   // already in exp2 domain (Q pre-scaled)
    }

    // ---- online softmax (per-query state is lane-local; partner = lane^16)
    float mloc = st[0][0];
#pragma unroll
    for (int t = 0; t < 4; ++t)
#pragma unroll
      for (int r = 0; r < 8; ++r) mloc = fmaxf(mloc, st[t][r]);
    mloc = fmaxf(mloc, __shfl_xor(mloc, 16, 32));

    const float m_new = fmaxf(m_run, mloc);
    const float corr  = exp2f(m_run - m_new);

    float p[4][8], rsum = 0.0f;
#pragma unroll
    for (int t = 0; t < 4; ++t)
#pragma unroll
      for (int r = 0; r < 8; ++r) {
        p[t][r] = exp2f(st[t][r] - m_new);
        rsum += p[t][r];
      }
    rsum += __shfl_xor(rsum, 16, 32);
    l_run = l_run * corr + rsum;
    m_run = m_new;

#pragma unroll
    for (int dt = 0; dt < 4; ++dt)
#pragma unroll
      for (int r = 0; r < 8; ++r) acc[dt][r] *= corr;

    // ---- P^T B-operands (keys 0-31 and 32-63) + V^T A-fragments, O^T accum.
    // V^T A elem e <-> key = 32c + e + 16h; row = d = dt*16 + qcol
#pragma unroll
    for (int c = 0; c < 2; ++c) {
      const v16bf pb = make_pb(p[2 * c], p[2 * c + 1], hi);
#pragma unroll
      for (int dt = 0; dt < 4; ++dt) {
        const __bf16* vr =
            &Vtls[buf][(dt * 16 + qcol) * kVStride + 32 * c + (hi ? 16 : 0)];
        const v16bf va = cat8(*reinterpret_cast<const v8bf*>(vr),
                              *reinterpret_cast<const v8bf*>(vr + 8));
        acc[dt] = __builtin_amdgcn_wmma_f32_16x16x32_bf16(false, va, false, pb,
                                                          (short)0, acc[dt],
                                                          false, false);
      }
    }

    // convert + store next chunk after compute; barrier publishes it
    if (more) stage_store((i + 1) & 1, nk, nv);
    __syncthreads();
  }

  // ---- epilogue: O = acc / l. Lane holds (query=qcol, d = dt*16 + r + 8h).
  const float invl = 1.0f / l_run;
  float* op = Out + ((size_t)batch * kLQ + qrow) * kD + (hi ? 8 : 0);
#pragma unroll
  for (int dt = 0; dt < 4; ++dt)
#pragma unroll
    for (int r = 0; r < 8; ++r)
      op[dt * 16 + r] = acc[dt][r] * invl;
}

extern "C" void kernel_launch(void* const* d_in, const int* in_sizes, int n_in,
                              void* d_out, int out_size, void* d_ws, size_t ws_size,
                              hipStream_t stream) {
  (void)in_sizes; (void)n_in; (void)out_size; (void)d_ws; (void)ws_size;
  const float* Q     = (const float*)d_in[0];
  const float* K     = (const float*)d_in[1];
  const float* V     = (const float*)d_in[2];
  const int*   scale = (const int*)d_in[3];   // python int scalar -> int[1]
  float*       Out   = (float*)d_out;

  dim3 grid(kLQ / 128, kB);   // (16, 16): 8 waves/block x 16 queries/wave
  fattn_bf16_wmma<<<grid, 256, 0, stream>>>(Q, K, V, scale, Out);
}
